// OFNFieldBuilder_7997229105586
// MI455X (gfx1250) — compile-verified, matrix operand order checked by reading, not run on hardware
//
#include <hip/hip_runtime.h>
#include <hip/hip_bf16.h>

// ---------------------------------------------------------------------------
// OFN field builder for MI455X (gfx1250, wave32):
//   1) updates = x @ W^T + b   (bf16 WMMA, f32 accum, async-to-LDS double buffer)
//   2) per-channel EMA scan along S via 3-stage chunked scan, in-place on d_out
// ---------------------------------------------------------------------------

#define D_MODEL 2048
#define N_OUT   2048           // SLOTS * D_FIELD
#define BATCH   2
#define SEQ     4096
#define M_TOTAL (BATCH * SEQ)  // 8192

#define KTILE 32
#define MTILE 128
#define NTILE 128
#define CHUNK 128              // scan chunk length
#define NCHUNK (SEQ / CHUNK)   // 32
#define NSEQ (BATCH * N_OUT)   // 4096 independent recurrences

typedef __bf16 bf16_t;
typedef __attribute__((ext_vector_type(16))) __bf16 v16bf;
typedef __attribute__((ext_vector_type(8)))  float  v8f;
typedef __attribute__((ext_vector_type(4)))  float  v4f;

// Matches the builtin's parameter type: vector_size(16) int, AS1/AS3 pointers.
typedef int v4i_vs __attribute__((__vector_size__(16)));
typedef __attribute__((address_space(1))) v4i_vs* gbl_v4i_ptr;
typedef __attribute__((address_space(3))) v4i_vs* lds_v4i_ptr;

// gfx1250 async global->LDS path (ASYNCcnt-tracked), with safe fallback.
#if defined(__AMDGCN__) && defined(__has_builtin)
#if __has_builtin(__builtin_amdgcn_global_load_async_to_lds_b128)
#define OFN_ASYNC 1
#endif
#endif
#ifndef OFN_ASYNC
#define OFN_ASYNC 0
#endif

template <int N>
__device__ __forceinline__ void wait_async() {
#if OFN_ASYNC
#if __has_builtin(__builtin_amdgcn_s_wait_asynccnt)
    __builtin_amdgcn_s_wait_asynccnt(N);
#else
    asm volatile("s_wait_asynccnt %0" ::"n"(N) : "memory");
#endif
#endif
}

// Each thread stages one 64B row-half of the A tile and one of the B tile.
__device__ __forceinline__ void stage_tiles(const float* __restrict__ xsrc,
                                            const float* __restrict__ wsrc,
                                            float* ldsA, float* ldsB) {
#if OFN_ASYNC
#pragma unroll
    for (int i = 0; i < 4; ++i) {
        __builtin_amdgcn_global_load_async_to_lds_b128(
            (gbl_v4i_ptr)(xsrc + i * 4), (lds_v4i_ptr)(ldsA + i * 4),
            /*offset=*/0, /*cpol=*/0);
        __builtin_amdgcn_global_load_async_to_lds_b128(
            (gbl_v4i_ptr)(wsrc + i * 4), (lds_v4i_ptr)(ldsB + i * 4),
            /*offset=*/0, /*cpol=*/0);
    }
#else
#pragma unroll
    for (int i = 0; i < 4; ++i) {
        v4f xa = *(const v4f*)(xsrc + i * 4);
        v4f wa = *(const v4f*)(wsrc + i * 4);
        *(v4f*)(ldsA + i * 4) = xa;
        *(v4f*)(ldsB + i * 4) = wa;
    }
#endif
}

// Build a 16-element bf16 WMMA fragment from 16 contiguous f32 in LDS.
__device__ __forceinline__ v16bf frag_from_f32(const float* __restrict__ p) {
    v4f a0 = *(const v4f*)(p + 0);
    v4f a1 = *(const v4f*)(p + 4);
    v4f a2 = *(const v4f*)(p + 8);
    v4f a3 = *(const v4f*)(p + 12);
    v16bf r;
#pragma unroll
    for (int i = 0; i < 4; ++i) {
        r[i + 0]  = (bf16_t)a0[i];
        r[i + 4]  = (bf16_t)a1[i];
        r[i + 8]  = (bf16_t)a2[i];
        r[i + 12] = (bf16_t)a3[i];
    }
    return r;
}

// ---------------------------------------------------------------------------
// Kernel 1: GEMM  out[m, n] = sum_k x[m,k] * W[n,k] + b[n]
// Block: 256 threads = 8 waves. Tile 128(M) x 128(N), K-step 32, double-buffered.
// Wave grid 4(M) x 2(N): each wave owns 32x64 = 2x4 WMMA accumulators.
// ---------------------------------------------------------------------------
__global__ __launch_bounds__(256) void ofn_gemm_bf16(
    const float* __restrict__ X, const float* __restrict__ W,
    const float* __restrict__ bias, float* __restrict__ out)
{
    __shared__ __align__(16) float AsF[2][MTILE * KTILE];  // 2 x 16 KB
    __shared__ __align__(16) float BsF[2][NTILE * KTILE];  // 2 x 16 KB

    const int bm = (blockIdx.x >> 4) * MTILE;   // 64 M-blocks
    const int bn = (blockIdx.x & 15) * NTILE;   // 16 N-blocks
    const int tid  = threadIdx.x;
    const int lane = tid & 31;
    const int wid  = tid >> 5;
    const int waveM = (wid >> 1) * 32;          // 0,32,64,96
    const int waveN = (wid & 1) * 64;           // 0,64

    v8f acc[2][4];
#pragma unroll
    for (int mi = 0; mi < 2; ++mi)
#pragma unroll
        for (int ni = 0; ni < 4; ++ni)
            acc[mi][ni] = (v8f)(0.0f);

    // Staging assignment: thread t owns row t>>1, 64B half (t&1) of each tile.
    const int lrow  = tid >> 1;          // 0..127
    const int lhalf = (tid & 1) * 16;    // 0 or 16
    const float* xbase = X + (size_t)(bm + lrow) * D_MODEL + lhalf;
    const float* wbase = W + (size_t)(bn + lrow) * D_MODEL + lhalf;
    const int lo = lrow * KTILE + lhalf;

    const int ar = lane & 15;            // row/col within 16-subtile
    const int ah = (lane >> 4) * 16;     // K-half select (contiguous k = 16h + s, same for A and B)

    constexpr int NSTEP = D_MODEL / KTILE;  // 64

    // Prologue: stage tile 0 into buffer 0.
    stage_tiles(xbase, wbase, &AsF[0][lo], &BsF[0][lo]);

    for (int it = 0; it < NSTEP; ++it) {
        const int cur = it & 1;
        if (it + 1 < NSTEP) {
            // Stream next K-tile into the other buffer while we compute.
            stage_tiles(xbase + (it + 1) * KTILE, wbase + (it + 1) * KTILE,
                        &AsF[cur ^ 1][lo], &BsF[cur ^ 1][lo]);
            wait_async<8>();   // current tile landed; next tile still in flight
        } else {
            wait_async<0>();
        }
        __syncthreads();

        const float* Ab = &AsF[cur][0];
        const float* Bb = &BsF[cur][0];
#pragma unroll
        for (int mi = 0; mi < 2; ++mi) {
            v16bf afrag = frag_from_f32(Ab + (waveM + mi * 16 + ar) * KTILE + ah);
#pragma unroll
            for (int ni = 0; ni < 4; ++ni) {
                v16bf bfrag = frag_from_f32(Bb + (waveN + ni * 16 + ar) * KTILE + ah);
                acc[mi][ni] = __builtin_amdgcn_wmma_f32_16x16x32_bf16(
                    /*neg_a=*/false, afrag, /*neg_b=*/false, bfrag,
                    /*c_mod=*/(short)0, acc[mi][ni],
                    /*reuse_a=*/false, /*reuse_b=*/false);
            }
        }
        __syncthreads();   // all waves done reading `cur` before it is restaged
    }

    // Epilogue: C/D layout (ISA 7.12.2): VGPR v -> M = v + 8*(lane>=16), N = lane&15
    const int row_in = (lane >> 4) * 8;
    const int col    = lane & 15;
#pragma unroll
    for (int mi = 0; mi < 2; ++mi) {
#pragma unroll
        for (int ni = 0; ni < 4; ++ni) {
            const int n = bn + waveN + ni * 16 + col;
            const float bv = bias[n];
            const int mbase = bm + waveM + mi * 16 + row_in;
#pragma unroll
            for (int v = 0; v < 8; ++v) {
                out[(size_t)(mbase + v) * N_OUT + n] = acc[mi][ni][v] + bv;
            }
        }
    }
}

// ---------------------------------------------------------------------------
// Scan stage A: per-(sequence, chunk) local carry (state at chunk end from 0).
// ---------------------------------------------------------------------------
__global__ __launch_bounds__(256) void ofn_scan_carry(
    const float* __restrict__ updates, const float* __restrict__ alphas,
    float* __restrict__ carry)
{
    const int idx   = blockIdx.x * 256 + threadIdx.x;   // 131072 threads
    const int ch    = idx & (N_OUT - 1);
    const int chunk = (idx >> 11) & (NCHUNK - 1);
    const int b     = idx >> 16;
    const float alpha = alphas[ch >> 8];                 // slot = ch / 256

    const float* p = updates + ((size_t)b * SEQ + (size_t)chunk * CHUNK) * N_OUT + ch;
    float c = 0.0f;
#pragma unroll 8
    for (int t = 0; t < CHUNK; ++t) {
        if ((t & 15) == 0 && t + 32 < CHUNK)
            __builtin_prefetch(p + (size_t)(t + 32) * N_OUT, 0, 0);
        c = fmaf(alpha, c, p[(size_t)t * N_OUT]);
    }
    carry[((size_t)b * N_OUT + ch) * NCHUNK + chunk] = c;
}

// ---------------------------------------------------------------------------
// Scan stage B: in-place exclusive scan of chunk carries per sequence.
// ---------------------------------------------------------------------------
__global__ __launch_bounds__(256) void ofn_scan_prefix(
    float* __restrict__ carry, const float* __restrict__ alphas)
{
    const int seq = blockIdx.x * 256 + threadIdx.x;      // 4096 threads
    const int ch  = seq & (N_OUT - 1);
    const float alpha = alphas[ch >> 8];
    const float aL = __powf(alpha, (float)CHUNK);
    float* c = carry + (size_t)seq * NCHUNK;
    float run = 0.0f;
#pragma unroll
    for (int j = 0; j < NCHUNK; ++j) {
        const float cj = c[j];
        c[j] = run;                 // exclusive prefix (state entering chunk j)
        run = fmaf(aL, run, cj);
    }
}

// ---------------------------------------------------------------------------
// Scan stage C: seeded local scan, in-place on d_out.
// ---------------------------------------------------------------------------
__global__ __launch_bounds__(256) void ofn_scan_apply(
    float* __restrict__ out, const float* __restrict__ alphas,
    const float* __restrict__ carry)
{
    const int idx   = blockIdx.x * 256 + threadIdx.x;    // 131072 threads
    const int ch    = idx & (N_OUT - 1);
    const int chunk = (idx >> 11) & (NCHUNK - 1);
    const int b     = idx >> 16;
    const float alpha = alphas[ch >> 8];

    float state = carry[((size_t)b * N_OUT + ch) * NCHUNK + chunk];
    float* p = out + ((size_t)b * SEQ + (size_t)chunk * CHUNK) * N_OUT + ch;
#pragma unroll 8
    for (int t = 0; t < CHUNK; ++t) {
        if ((t & 15) == 0 && t + 32 < CHUNK)
            __builtin_prefetch(p + (size_t)(t + 32) * N_OUT, 0, 0);
        const float u = p[(size_t)t * N_OUT];
        state = fmaf(alpha, state, u);
        p[(size_t)t * N_OUT] = state;
    }
}

// ---------------------------------------------------------------------------
extern "C" void kernel_launch(void* const* d_in, const int* in_sizes, int n_in,
                              void* d_out, int out_size, void* d_ws, size_t ws_size,
                              hipStream_t stream)
{
    const float* x      = (const float*)d_in[0];
    const float* W      = (const float*)d_in[1];
    const float* bias   = (const float*)d_in[2];
    const float* alphas = (const float*)d_in[3];
    float* out   = (float*)d_out;
    float* carry = (float*)d_ws;   // needs NSEQ * NCHUNK * 4 = 512 KB

    // 1) GEMM: updates -> d_out
    ofn_gemm_bf16<<<(M_TOTAL / MTILE) * (N_OUT / NTILE), 256, 0, stream>>>(x, W, bias, out);
    // 2) chunked EMA scan, in place on d_out
    const int nthreads = NSEQ * NCHUNK;                  // 131072
    ofn_scan_carry <<<nthreads / 256, 256, 0, stream>>>(out, alphas, carry);
    ofn_scan_prefix<<<NSEQ / 256,     256, 0, stream>>>(carry, alphas);
    ofn_scan_apply <<<nthreads / 256, 256, 0, stream>>>(out, alphas, carry);
}
